// LSTM_one_54202487276118
// MI455X (gfx1250) — matrix operand order repaired and checked by reading, not compiled, not run on hardware
//
#include <hip/hip_runtime.h>
#include <hip/hip_bf16.h>
#include <hip/hip_fp16.h>

#define N_CLASS 32000
#define EMB     512
#define HID     1024
#define BATCH   64
#define SEQ     256
#define WROW    (HID+EMB)        // 1536: weight row, [0,1024)=h part, [1024,1536)=x part
#define NBLK    64               // persistent blocks, one 16-wide hidden tile each

typedef __attribute__((ext_vector_type(16))) _Float16 v16h;
typedef __attribute__((ext_vector_type(8)))  float    v8f;
typedef __attribute__((ext_vector_type(4)))  float    v4f;
typedef __attribute__((ext_vector_type(4)))  unsigned v4u;

union U16H { v16h h; v4u u[2]; _Float16 e[16]; };
union U8F  { v8f  f; v4f v[2]; float    e[8];  };
union U4H  { unsigned long long u; _Float16 e[4]; };

__device__ inline v8f wmma16(v16h a, v16h b, v8f c) {
  // v_wmma_f32_16x16x32_f16: (neg_a, A, neg_b, B, c_mod, C, reuse_a, reuse_b)
  return __builtin_amdgcn_wmma_f32_16x16x32_f16(false, a, false, b, (short)0, c, false, false);
}

__device__ inline v8f splat8(float x) {
  U8F o;
#pragma unroll
  for (int r = 0; r < 8; ++r) o.e[r] = x;
  return o.f;
}

// 16 consecutive f32 -> 16 packed f16 (one B-operand lane's worth)
__device__ inline v16h cvt16(const float* p) {
  v4f f0 = *(const v4f*)(p + 0);
  v4f f1 = *(const v4f*)(p + 4);
  v4f f2 = *(const v4f*)(p + 8);
  v4f f3 = *(const v4f*)(p + 12);
  U16H o;
#pragma unroll
  for (int i = 0; i < 4; ++i) {
    o.e[i]      = (_Float16)f0[i];
    o.e[4 + i]  = (_Float16)f1[i];
    o.e[8 + i]  = (_Float16)f2[i];
    o.e[12 + i] = (_Float16)f3[i];
  }
  return o.h;
}

// WMMA A operand (16x32 f16): lane l<16 -> row rowbase+l, K k0..k0+15 contiguous;
// lane l>=16 -> same rows, K k0+16..k0+31. Source is row-major f16 [rows][stride].
__device__ inline v16h load_a(const _Float16* base, int stride, int rowbase, int k0, int lane) {
  const _Float16* p = base + (size_t)(rowbase + (lane & 15)) * stride + k0 + ((lane & 16) ? 16 : 0);
  U16H a;
  a.u[0] = *(const v4u*)(p);
  a.u[1] = *(const v4u*)(p + 8);
  return a.h;
}

__device__ inline float sigm(float x) { return 1.0f / (1.0f + __expf(-x)); }

// ---------------------------------------------------------------------------
// Kernel 1: gather + convert embeddings: xs[t][b][e] (f16) = emb[X[b][t]][e]
// ---------------------------------------------------------------------------
__global__ __launch_bounds__(256) void k_embed(
    const int* __restrict__ X, const float* __restrict__ emb,
    _Float16* __restrict__ xs)
{
  const int idx = (blockIdx.x * 256 + threadIdx.x) * 4;  // 4 elements/thread
  const int t   = idx >> 15;            // / (64*512)
  const int rem = idx & 32767;
  const int b   = rem >> 9;
  const int e   = rem & 511;
  const int id  = X[b * SEQ + t];
  v4f v = *(const v4f*)(emb + (size_t)id * EMB + e);
  U4H o;
#pragma unroll
  for (int i = 0; i < 4; ++i) o.e[i] = (_Float16)v[i];
  *(unsigned long long*)(xs + idx) = o.u;
}

// ---------------------------------------------------------------------------
// Kernel 2: prepack x-part gate weights into f16 WMMA-B lane layout.
// Wxp[tile][kt][lane][16 halves]; tile = global 16-col tile of the 4096 gate
// outputs (gate-major), kt in [0,16) covers K=EMB, lane entry = 16 consecutive
// k values of row col (x columns 1024..1535 of the weight row).
// ---------------------------------------------------------------------------
__global__ __launch_bounds__(256) void k_packwx(
    const float* __restrict__ Wf, const float* __restrict__ Wi,
    const float* __restrict__ Wc, const float* __restrict__ Wo,
    _Float16* __restrict__ Wxp)
{
  const int task = blockIdx.x * 256 + threadIdx.x;  // 256 tiles * 16 kt * 32 lanes
  const int tile = task >> 9;
  const int kt   = (task >> 5) & 15;
  const int l    = task & 31;
  const int col  = tile * 16 + (l & 15);
  const int gate = col >> 10;
  const int j    = col & (HID - 1);
  const float* Wg = (gate == 0) ? Wf : (gate == 1) ? Wi : (gate == 2) ? Wc : Wo;
  const float* p  = Wg + (size_t)j * WROW + HID + kt * 32 + ((l & 16) ? 16 : 0);
  U16H vb; vb.h = cvt16(p);
  v4u* dst = (v4u*)(Wxp + (size_t)task * 16);
  dst[0] = vb.u[0];
  dst[1] = vb.u[1];
}

// ---------------------------------------------------------------------------
// Kernel 3: persistent recurrent kernel. 64 blocks x 256 threads (8 waves).
// Block bk owns hidden cols [16*bk, 16*bk+16): all 4 gates, all 4 batch tiles.
// h-part weights (128 KB f16) are LDS-resident in WMMA-B layout; x-part B tiles
// stream from the L2-resident Wxp pack. Device-wide split barrier per step; the
// step-(t+1) x-contribution is computed in the barrier's latency shadow.
// ---------------------------------------------------------------------------
__global__ __launch_bounds__(256) void k_lstm(
    const float* __restrict__ Wf, const float* __restrict__ Wi,
    const float* __restrict__ Wc, const float* __restrict__ Wo,
    const float* __restrict__ bf, const float* __restrict__ bi,
    const float* __restrict__ bc, const float* __restrict__ bo,
    const _Float16* __restrict__ xs, const _Float16* __restrict__ Wxp,
    _Float16* __restrict__ hA, _Float16* __restrict__ hB,
    float* __restrict__ out_h, float* __restrict__ out_c,
    unsigned* __restrict__ bar)
{
  __shared__ _Float16 ldsB[4 * 32 * 32 * 16]; // [gate][ktile][lane][16 halves] = 128 KB
  __shared__ float    ldsG[4 * 16 * BATCH];   // activated gates [gate][col][batch] = 16 KB

  const int tid     = threadIdx.x;
  const int lane    = tid & 31;
  const int w       = tid >> 5;
  const int bk      = blockIdx.x;
  const int colbase = bk * 16;

  // one-time pack: h-part W[col][k] (f32) -> f16 WMMA-B per-lane layout in LDS
  for (int task = tid; task < 4096; task += 256) {
    const int g  = task >> 10;
    const int kt = (task >> 5) & 31;
    const int l  = task & 31;
    const float* Wg = (g == 0) ? Wf : (g == 1) ? Wi : (g == 2) ? Wc : Wo;
    const float* p  = Wg + (size_t)(colbase + (l & 15)) * WROW + kt * 32 + ((l & 16) ? 16 : 0);
    U16H vb; vb.h = cvt16(p);
    v4u* dst = (v4u*)(ldsB + (size_t)task * 16);
    dst[0] = vb.u[0];
    dst[1] = vb.u[1];
  }
  __syncthreads();

  const int g   = w >> 1;                       // gate for this wave
  const int mA  = (w & 1) * 2;                  // two batch tiles per wave
  const int mB  = mA + 1;
  const int b0A = mA * 16 + ((lane & 16) ? 8 : 0);
  const int b0B = mB * 16 + ((lane & 16) ? 8 : 0);

  const float* bgp = (g == 0) ? bf : (g == 1) ? bi : (g == 2) ? bc : bo;
  const float  bias = bgp[colbase + (lane & 15)];
  const _Float16* wxp = Wxp + ((size_t)(g * 64 + bk) * 512 + lane) * 16;

  float creg[4] = {0.f, 0.f, 0.f, 0.f};        // cell state, fixed (b,col) map per thread

  // x-part of step t: xacc = bias + x_t . Wx^T  (no cross-block dependence)
  auto xpart = [&](int t, v8f& x0, v8f& x1) {
    x0 = splat8(bias);
    x1 = splat8(bias);
    const _Float16* xs_t = xs + (size_t)t * BATCH * EMB;
    for (int kt = 0; kt < EMB / 32; ++kt) {
      U16H bm;
      const v4u* bp = (const v4u*)(wxp + (size_t)kt * 32 * 16);
      bm.u[0] = bp[0]; bm.u[1] = bp[1];
      const int k0 = kt * 32;
      v16h a0 = load_a(xs_t, EMB, mA * 16, k0, lane);
      v16h a1 = load_a(xs_t, EMB, mB * 16, k0, lane);
      x0 = wmma16(a0, bm.h, x0);
      x1 = wmma16(a1, bm.h, x1);
    }
  };

  v8f xacc0, xacc1;
  xpart(0, xacc0, xacc1);
  unsigned pend = 0xFFFFFFFFu;                 // sentinel: nothing to wait on (t=0)

  for (int t = 0; t < SEQ; ++t) {
    // ---- barrier WAIT phase (h_t now visible from all blocks) ----
    if (tid == 0 && pend != 0xFFFFFFFFu) {
      while (__hip_atomic_load(&bar[1], __ATOMIC_ACQUIRE, __HIP_MEMORY_SCOPE_AGENT) == pend)
        __builtin_amdgcn_s_sleep(1);
    }
    __syncthreads();

    const _Float16* hin  = (t & 1) ? hB : hA;
    _Float16*       hout = (t & 1) ? hA : hB;

    // ---- h-part: gates += h_t . Wh^T (K=1024, B tiles from LDS) ----
    v8f acc0 = xacc0, acc1 = xacc1;
    for (int kt = 0; kt < HID / 32; ++kt) {
      const int k0 = kt * 32;
      U16H bm;
      const v4u* bp = (const v4u*)(ldsB + (size_t)((g * 32 + kt) * 32 + lane) * 16);
      bm.u[0] = bp[0]; bm.u[1] = bp[1];
      v16h a0 = load_a(hin, HID, mA * 16, k0, lane);
      v16h a1 = load_a(hin, HID, mB * 16, k0, lane);
      acc0 = wmma16(a0, bm.h, acc0);
      acc1 = wmma16(a1, bm.h, acc1);
    }

    // ---- activations + stage gates to LDS [gate][col][batch] ----
    {
      U8F u0, u1; u0.f = acc0; u1.f = acc1;
#pragma unroll
      for (int r = 0; r < 8; ++r) {
        u0.e[r] = (g == 2) ? tanhf(u0.e[r]) : sigm(u0.e[r]);
        u1.e[r] = (g == 2) ? tanhf(u1.e[r]) : sigm(u1.e[r]);
      }
      float* gp = ldsG + (g * 16 + (lane & 15)) * BATCH;
      *(v4f*)(gp + b0A)     = u0.v[0];
      *(v4f*)(gp + b0A + 4) = u0.v[1];
      *(v4f*)(gp + b0B)     = u1.v[0];
      *(v4f*)(gp + b0B + 4) = u1.v[1];
    }
    __syncthreads();

    // ---- elementwise cell update: 4 (b,col) elements per thread ----
#pragma unroll
    for (int q = 0; q < 4; ++q) {
      const int e   = tid * 4 + q;
      const int col = e & 15;
      const int b   = e >> 4;
      const float fg = ldsG[(0 * 16 + col) * BATCH + b];
      const float ig = ldsG[(1 * 16 + col) * BATCH + b];
      const float cc = ldsG[(2 * 16 + col) * BATCH + b];
      const float og = ldsG[(3 * 16 + col) * BATCH + b];
      const float cn = fg * creg[q] + ig * cc;
      creg[q] = cn;
      const float hn = og * tanhf(cn);
      hout[(size_t)b * HID + colbase + col] = (_Float16)hn;
      if (t == SEQ - 1) {
        out_h[(size_t)b * HID + colbase + col] = hn;
        out_c[(size_t)b * HID + colbase + col] = cn;
      }
    }

    // ---- barrier ARRIVE phase ----
    __threadfence();
    __syncthreads();
    if (tid == 0) {
      unsigned gen = __hip_atomic_load(&bar[1], __ATOMIC_RELAXED, __HIP_MEMORY_SCOPE_AGENT);
      unsigned n   = __hip_atomic_fetch_add(&bar[0], 1u, __ATOMIC_ACQ_REL, __HIP_MEMORY_SCOPE_AGENT);
      if (n == NBLK - 1) {
        __hip_atomic_store(&bar[0], 0u, __ATOMIC_RELAXED, __HIP_MEMORY_SCOPE_AGENT);
        __hip_atomic_fetch_add(&bar[1], 1u, __ATOMIC_ACQ_REL, __HIP_MEMORY_SCOPE_AGENT);
        pend = 0xFFFFFFFFu;                    // we were last: nothing to wait on
      } else {
        pend = gen;
      }
    }

    // ---- hide step-(t+1) x-part GEMM in the barrier latency shadow ----
    if (t + 1 < SEQ) xpart(t + 1, xacc0, xacc1);
  }
}

// ---------------------------------------------------------------------------
// Kernel 4: logits = h_final . W^T + b  (64 x 1024 x 32000), W f32->f16 on the
// fly per tile. One wave per 16-col tile, 4 batch tiles each.
// ---------------------------------------------------------------------------
__global__ __launch_bounds__(256) void k_logits(
    const _Float16* __restrict__ hA, const float* __restrict__ W,
    const float* __restrict__ bias, float* __restrict__ out)
{
  const int lane = threadIdx.x & 31;
  const int gw   = blockIdx.x * 8 + (threadIdx.x >> 5);
  if (gw >= N_CLASS / 16) return;
  const int col = gw * 16 + (lane & 15);
  const float* wrow = W + (size_t)col * HID + ((lane & 16) ? 16 : 0);
  const float  bv   = bias[col];

  v8f acc[4];
#pragma unroll
  for (int m = 0; m < 4; ++m) acc[m] = splat8(bv);

  for (int k0 = 0; k0 < HID; k0 += 32) {
    v16h bm = cvt16(wrow + k0);
#pragma unroll
    for (int m = 0; m < 4; ++m) {
      v16h a = load_a(hA, HID, m * 16, k0, lane);
      acc[m] = wmma16(a, bm, acc[m]);
    }
  }

  const int rbase = (lane & 16) ? 8 : 0;
#pragma unroll
  for (int m = 0; m < 4; ++m) {
    U8F u; u.f = acc[m];
#pragma unroll
    for (int r = 0; r < 8; ++r)
      out[(size_t)(m * 16 + rbase + r) * N_CLASS + col] = u.e[r];
  }
}

// ---------------------------------------------------------------------------
extern "C" void kernel_launch(void* const* d_in, const int* in_sizes, int n_in,
                              void* d_out, int out_size, void* d_ws, size_t ws_size,
                              hipStream_t stream) {
  (void)in_sizes; (void)n_in; (void)out_size; (void)ws_size;

  const int*   X   = (const int*)  d_in[0];
  const float* emb = (const float*)d_in[1];
  const float* Wf  = (const float*)d_in[2];
  const float* bf  = (const float*)d_in[3];
  const float* Wi  = (const float*)d_in[4];
  const float* bi  = (const float*)d_in[5];
  const float* Wc  = (const float*)d_in[6];
  const float* bc  = (const float*)d_in[7];
  const float* Wo  = (const float*)d_in[8];
  const float* bo  = (const float*)d_in[9];
  const float* W   = (const float*)d_in[10];
  const float* b   = (const float*)d_in[11];

  float* out_logits = (float*)d_out;                    // [64][32000]
  float* out_h      = out_logits + (size_t)BATCH * N_CLASS;
  float* out_c      = out_h + (size_t)BATCH * HID;

  // workspace (~22 MB): [0,256) barrier | 4K: hA,hB (128K each) | 512K: xs f16
  // (16.8 MB) | 20M: Wxp f16 (4 MB)
  char*     ws  = (char*)d_ws;
  unsigned* bar = (unsigned*)ws;
  _Float16* hA  = (_Float16*)(ws + 4096);
  _Float16* hB  = hA + (size_t)BATCH * HID;
  _Float16* xs  = (_Float16*)(ws + (1u << 19));
  _Float16* Wxp = (_Float16*)(ws + (20u << 20));

  hipMemsetAsync(bar, 0, 256, stream);                                    // barrier state
  hipMemsetAsync(hA, 0, (size_t)BATCH * HID * sizeof(_Float16), stream);  // h0 = 0

  k_embed <<<dim3(8192), dim3(256), 0, stream>>>(X, emb, xs);
  k_packwx<<<dim3(512),  dim3(256), 0, stream>>>(Wf, Wi, Wc, Wo, Wxp);
  k_lstm  <<<dim3(NBLK), dim3(256), 0, stream>>>(Wf, Wi, Wc, Wo, bf, bi, bc, bo,
                                                 xs, Wxp, hA, hB, out_h, out_c, bar);
  // 256 steps: final h (f16) ends in hA (t=255 odd writes hA)
  k_logits<<<dim3(N_CLASS / 16 / 8), dim3(256), 0, stream>>>(hA, W, b, out_logits);
}
